// GNNModel_10703058501976
// MI455X (gfx1250) — matrix operand order repaired and checked by reading, not compiled, not run on hardware
//
#include <hip/hip_runtime.h>
#include <hip/hip_bf16.h>

typedef __attribute__((ext_vector_type(16))) __bf16 v16bf;
typedef __attribute__((ext_vector_type(8)))  float  v8f;

// ---------------------------------------------------------------------------
// Degree / normalization precompute
// ---------------------------------------------------------------------------
__global__ void zero_i32_k(int* __restrict__ p, int n) {
    int i = blockIdx.x * blockDim.x + threadIdx.x;
    if (i < n) p[i] = 0;
}

__global__ void count_deg_k(const int* __restrict__ dst, int* __restrict__ deg, int nE) {
    int e = blockIdx.x * blockDim.x + threadIdx.x;
    if (e < nE) atomicAdd(&deg[dst[e]], 1);
}

__global__ void dinv_k(const int* __restrict__ deg, float* __restrict__ dinv, int n) {
    int i = blockIdx.x * blockDim.x + threadIdx.x;
    if (i < n) dinv[i] = rsqrtf((float)deg[i] + 1.0f);
}

__global__ void edge_norm_k(const int* __restrict__ src, const int* __restrict__ dst,
                            const float* __restrict__ dinv, float* __restrict__ enorm, int nE) {
    int e = blockIdx.x * blockDim.x + threadIdx.x;
    if (e < nE) enorm[e] = dinv[src[e]] * dinv[dst[e]];
}

// ---------------------------------------------------------------------------
// W pre-pass: f32 row-major [din,dout] -> bf16 col-major K-padded [dout,kp],
// kp = max(din,32), tail zero-filled. Makes the GEMM B-operand a single
// contiguous 32B load per lane per k-step, with no bounds checks.
// ---------------------------------------------------------------------------
__global__ void wconv_k(const float* __restrict__ W, __bf16* __restrict__ Wt,
                        int din, int dout, int kp) {
    int tid = blockIdx.x * blockDim.x + threadIdx.x;
    if (tid >= dout * kp) return;
    int c = tid / kp, k = tid - c * kp;
    float f = (k < din) ? W[k * dout + c] : 0.0f;
    Wt[tid] = (__bf16)f;
}

// ---------------------------------------------------------------------------
// Dense GEMM: C[N,DOUT] = (relu?)(A[N,DIN]) @ W[DIN,DOUT], bf16 WMMA, f32 acc.
// One wave32 computes TN adjacent 16x16 output tiles (A-register reuse).
// A-operand layout (wave32): lane = row (l15), kh = lane>>4;
//   elements e -> K = k0 + 16*kh + e   (contiguous, 4x float4)
// B-operand: lane = col; elements e -> K = k0 + 16*kh + e  -> one v16bf load
// from the pre-converted col-major bf16 Wt.
// C/D: VGPR r -> row r + 8*kh, col l15.
// ---------------------------------------------------------------------------
template<int DIN, int DOUT, bool RELU>
__global__ void gemm_wmma_k(const float* __restrict__ A, const __bf16* __restrict__ Wt,
                            float* __restrict__ C, int nrows) {
    constexpr int KP  = (DIN < 32) ? 32 : DIN;    // padded K stride of Wt
    constexpr int TN  = (DOUT >= 32) ? 2 : 1;     // col tiles per wave
    constexpr int TGN = DOUT / (16 * TN);         // col-tile groups
    int wave = (int)((blockIdx.x * blockDim.x + threadIdx.x) >> 5);
    int lane = threadIdx.x & 31;
    int tiles_m = (nrows + 15) >> 4;
    if (wave >= tiles_m * TGN) return;
    int tm = wave / TGN, tg = wave % TGN;
    int row0 = tm << 4;
    int kh  = lane >> 4;
    int l15 = lane & 15;
    int arow = row0 + l15;
    int arowc = (arow < nrows) ? arow : (nrows - 1);   // clamp, no branch
    const float* arp = A + (long)arowc * DIN;

    v8f acc[TN];
    v8f zero = {};
    #pragma unroll
    for (int t = 0; t < TN; ++t) acc[t] = zero;

    #pragma unroll
    for (int k0 = 0; k0 < DIN; k0 += 32) {
        // ---- A operand: 4x float4 contiguous loads + pack to bf16 ----
        float av[16];
        if constexpr (DIN % 32 == 0) {
            const float* p0 = arp + k0 + 16 * kh;
            const float* p1 = arp + k0 + 16 * kh + 8;
            float4 q0 = *(const float4*)(p0);
            float4 q1 = *(const float4*)(p0 + 4);
            float4 q2 = *(const float4*)(p1);
            float4 q3 = *(const float4*)(p1 + 4);
            av[0]=q0.x; av[1]=q0.y; av[2]=q0.z; av[3]=q0.w;
            av[4]=q1.x; av[5]=q1.y; av[6]=q1.z; av[7]=q1.w;
            av[8]=q2.x; av[9]=q2.y; av[10]=q2.z; av[11]=q2.w;
            av[12]=q3.x; av[13]=q3.y; av[14]=q3.z; av[15]=q3.w;
        } else {
            // DIN == 16: kh=0 holds K=0..15, kh=1 (K>=16) is zero padding
            const float* p0 = arp;
            float4 q0 = *(const float4*)(p0);
            float4 q1 = *(const float4*)(p0 + 4);
            float4 q2 = *(const float4*)(p0 + 8);
            float4 q3 = *(const float4*)(p0 + 12);
            av[0]=q0.x; av[1]=q0.y; av[2]=q0.z; av[3]=q0.w;
            av[4]=q1.x; av[5]=q1.y; av[6]=q1.z; av[7]=q1.w;
            av[8]=q2.x; av[9]=q2.y; av[10]=q2.z; av[11]=q2.w;
            av[12]=q3.x; av[13]=q3.y; av[14]=q3.z; av[15]=q3.w;
            #pragma unroll
            for (int e = 0; e < 16; ++e) if (kh) av[e] = 0.0f;
        }
        v16bf a;
        #pragma unroll
        for (int e = 0; e < 16; ++e) {
            float f = av[e];
            if constexpr (RELU) f = fmaxf(f, 0.0f);
            a[e] = (__bf16)f;
        }
        // ---- B operand: one contiguous 32B bf16 load per col tile + WMMA ----
        #pragma unroll
        for (int t = 0; t < TN; ++t) {
            int bcol = (tg * TN + t) * 16 + l15;
            v16bf b = *(const v16bf*)(Wt + (long)bcol * KP + k0 + 16 * kh);
            acc[t] = __builtin_amdgcn_wmma_f32_16x16x32_bf16(
                         false, a, false, b, (short)0, acc[t], false, false);
        }
    }
    #pragma unroll
    for (int t = 0; t < TN; ++t) {
        int bcol = (tg * TN + t) * 16 + l15;
        #pragma unroll
        for (int r = 0; r < 8; ++r) {
            int row = row0 + r + 8 * kh;
            if (row < nrows) C[(long)row * DOUT + bcol] = acc[t][r];
        }
    }
}

// ---------------------------------------------------------------------------
// hout = hW * dinv[i]^2 + bias[d]   (self-loop term + bias; base for scatter)
// ---------------------------------------------------------------------------
__global__ void self_bias_k(const float* __restrict__ hW, const float* __restrict__ dinv,
                            const float* __restrict__ bias, float* __restrict__ hout,
                            int n, int dout) {
    int tid = blockIdx.x * blockDim.x + threadIdx.x;
    if (tid >= n * dout) return;
    int i = tid / dout, d = tid - i * dout;
    float dv = dinv[i];
    hout[tid] = hW[tid] * dv * dv + bias[d];
}

// ---------------------------------------------------------------------------
// Edge scatter-add: hout[dst] += hW[src] * enorm, float4 per thread.
// hW rows are L2-resident (<=102MB << 192MB L2).
// ---------------------------------------------------------------------------
__global__ void scatter_k(const float* __restrict__ hW, const int* __restrict__ src,
                          const int* __restrict__ dst, const float* __restrict__ enorm,
                          float* __restrict__ hout, int nE, int dout) {
    int per_edge = dout >> 2;
    int tid = blockIdx.x * blockDim.x + threadIdx.x;
    int e = tid / per_edge;
    if (e >= nE) return;
    int q = (tid - e * per_edge) << 2;
    int s = src[e], d = dst[e];
    float w = enorm[e];
    if (e + 1 < nE) __builtin_prefetch(&hW[(long)src[e + 1] * dout + q], 0, 1);
    float4 v = *(const float4*)(hW + (long)s * dout + q);
    float* dp = hout + (long)d * dout + q;
    unsafeAtomicAdd(dp + 0, v.x * w);
    unsafeAtomicAdd(dp + 1, v.y * w);
    unsafeAtomicAdd(dp + 2, v.z * w);
    unsafeAtomicAdd(dp + 3, v.w * w);
}

// ---------------------------------------------------------------------------
// TopK pooling
// ---------------------------------------------------------------------------
__global__ void pw_invnorm_k(const float* __restrict__ pw, int dim, float* __restrict__ out) {
    __shared__ float s[256];
    float v = (threadIdx.x < (unsigned)dim) ? pw[threadIdx.x] : 0.0f;
    s[threadIdx.x] = v * v;
    __syncthreads();
    for (int st = 128; st > 0; st >>= 1) {
        if ((int)threadIdx.x < st) s[threadIdx.x] += s[threadIdx.x + st];
        __syncthreads();
    }
    if (threadIdx.x == 0) out[0] = rsqrtf(s[0]);
}

__global__ void score_k(const float* __restrict__ h, const float* __restrict__ pw,
                        const float* __restrict__ invn, float* __restrict__ score,
                        int n, int dim) {
    int i = blockIdx.x * blockDim.x + threadIdx.x;
    if (i >= n) return;
    float acc = 0.0f;
    const float* hp = h + (long)i * dim;
    for (int d = 0; d < dim; d += 4) {
        float4 v = *(const float4*)(hp + d);
        float4 w = *(const float4*)(pw + d);
        acc += v.x * w.x + v.y * w.y + v.z * w.z + v.w * w.w;
    }
    score[i] = tanhf(acc * invn[0]);
}

// Pack (score, index) so that descending uint64 order == jax top_k order
// (higher score first; ties broken by lower index).
__global__ void keys_init_k(const float* __restrict__ score, unsigned long long* __restrict__ keys,
                            int n, int npad) {
    int i = blockIdx.x * blockDim.x + threadIdx.x;
    if (i >= npad) return;
    if (i < n) {
        unsigned u = __builtin_bit_cast(unsigned, score[i]);
        unsigned ord = (u & 0x80000000u) ? ~u : (u | 0x80000000u);
        keys[i] = ((unsigned long long)ord << 32) | (unsigned long long)(0xFFFFFFFFu - (unsigned)i);
    } else {
        keys[i] = 0ull;   // -inf pad, never reaches top-k since k <= n
    }
}

__global__ void bitonic_step_k(unsigned long long* __restrict__ keys, int j, int k) {
    int i = blockIdx.x * blockDim.x + threadIdx.x;
    int ixj = i ^ j;
    if (ixj <= i) return;
    unsigned long long a = keys[i], b = keys[ixj];
    bool descending_region = ((i & k) == 0);
    bool swap = descending_region ? (a < b) : (a > b);
    if (swap) { keys[i] = b; keys[ixj] = a; }
}

__global__ void gather_out_k(const unsigned long long* __restrict__ keys,
                             const float* __restrict__ h, const float* __restrict__ score,
                             float* __restrict__ out, int kpool, int dim) {
    int tid = blockIdx.x * blockDim.x + threadIdx.x;
    if (tid >= kpool * dim) return;
    int kth = tid / dim, d = tid - kth * dim;
    unsigned idx = 0xFFFFFFFFu - (unsigned)(keys[kth] & 0xFFFFFFFFull);
    out[tid] = h[(long)idx * dim + d] * score[idx];
}

// ---------------------------------------------------------------------------
// Launch
// ---------------------------------------------------------------------------
static inline size_t align256(size_t x) { return (x + 255) & ~(size_t)255; }

extern "C" void kernel_launch(void* const* d_in, const int* in_sizes, int n_in,
                              void* d_out, int out_size, void* d_ws, size_t ws_size,
                              hipStream_t stream) {
    const int dims[11] = {128, 256, 128, 64, 32, 16, 32, 64, 128, 256, 128};
    const float* x = (const float*)d_in[0];
    const int*  ei = (const int*)d_in[1];
    const int E = in_sizes[1] / 2;
    const int* src = ei;
    const int* dst = ei + E;
    const int N = in_sizes[0] / dims[0];
    const float* Wp[10]; const float* bp[10];
    for (int i = 0; i < 10; ++i) {
        Wp[i] = (const float*)d_in[2 + 2 * i];
        bp[i] = (const float*)d_in[3 + 2 * i];
    }
    const float* pw = (const float*)d_in[2 + 2 * 10];
    const int KPOOL = out_size / dims[10];

    // workspace carve-up
    char* w = (char*)d_ws;
    size_t BUF = align256((size_t)N * 256 * sizeof(float));
    float* buf[3];
    buf[0] = (float*)w; w += BUF;
    buf[1] = (float*)w; w += BUF;
    buf[2] = (float*)w; w += BUF;
    int*   deg   = (int*)w;   w += align256((size_t)N * 4);
    float* dinv  = (float*)w; w += align256((size_t)N * 4);
    float* enorm = (float*)w; w += align256((size_t)E * 4);
    float* score = (float*)w; w += align256((size_t)N * 4);
    float* invn  = (float*)w; w += 256;
    __bf16* Wt   = (__bf16*)w; w += align256((size_t)256 * 256 * sizeof(__bf16));
    int NPAD = 1; while (NPAD < N) NPAD <<= 1;
    unsigned long long* keys = (unsigned long long*)w; w += align256((size_t)NPAD * 8);

    const int B = 256;
    // --- degree / norms ---
    zero_i32_k<<<(N + B - 1) / B, B, 0, stream>>>(deg, N);
    count_deg_k<<<(E + B - 1) / B, B, 0, stream>>>(dst, deg, E);
    dinv_k<<<(N + B - 1) / B, B, 0, stream>>>(deg, dinv, N);
    edge_norm_k<<<(E + B - 1) / B, B, 0, stream>>>(src, dst, dinv, enorm, E);

    // --- 10 GCN layers ---
    const int tiles_m = (N + 15) >> 4;
#define GEMM_LAUNCH(DIN_, DOUT_, RELU_)                                           \
    do {                                                                          \
        constexpr int kp_  = (DIN_ < 32) ? 32 : DIN_;                             \
        int wth_ = DOUT_ * kp_;                                                   \
        wconv_k<<<(wth_ + B - 1) / B, B, 0, stream>>>(Wp[L], Wt, DIN_, DOUT_, kp_); \
        constexpr int tgn_ = (DOUT_ >= 32) ? (DOUT_ / 32) : (DOUT_ / 16);         \
        long th_ = (long)tiles_m * tgn_ * 32;                                     \
        gemm_wmma_k<DIN_, DOUT_, RELU_>                                           \
            <<<(int)((th_ + B - 1) / B), B, 0, stream>>>(hin, Wt, hW, N);         \
    } while (0)

    const float* hin = x;
    int hwIdx = 0, outIdx = 1;
    for (int L = 0; L < 10; ++L) {
        int dout = dims[L + 1];
        float* hW   = buf[hwIdx];
        float* hout = buf[outIdx];
        switch (L) {
            case 0: GEMM_LAUNCH(128, 256, false); break;
            case 1: GEMM_LAUNCH(256, 128, true);  break;
            case 2: GEMM_LAUNCH(128, 64,  true);  break;
            case 3: GEMM_LAUNCH(64,  32,  true);  break;
            case 4: GEMM_LAUNCH(32,  16,  true);  break;
            case 5: GEMM_LAUNCH(16,  32,  true);  break;
            case 6: GEMM_LAUNCH(32,  64,  true);  break;
            case 7: GEMM_LAUNCH(64,  128, true);  break;
            case 8: GEMM_LAUNCH(128, 256, true);  break;
            case 9: GEMM_LAUNCH(256, 128, true);  break;
        }
        long nelem = (long)N * dout;
        self_bias_k<<<(int)((nelem + B - 1) / B), B, 0, stream>>>(hW, dinv, bp[L], hout, N, dout);
        long sthreads = (long)E * (dout >> 2);
        scatter_k<<<(int)((sthreads + B - 1) / B), B, 0, stream>>>(hW, src, dst, enorm, hout, E, dout);
        // rotate: next layer reads hout; the other two slabs become hW/out
        hin = hout;
        int inIdx = outIdx;
        hwIdx  = (inIdx + 1) % 3;
        outIdx = (inIdx + 2) % 3;
    }
#undef GEMM_LAUNCH

    // --- TopK pooling ---
    const float* hfin = hin;                     // [N, 128], no ReLU on last layer
    pw_invnorm_k<<<1, 256, 0, stream>>>(pw, dims[10], invn);
    score_k<<<(N + B - 1) / B, B, 0, stream>>>(hfin, pw, invn, score, N, dims[10]);
    keys_init_k<<<(NPAD + B - 1) / B, B, 0, stream>>>(score, keys, N, NPAD);
    for (int k = 2; k <= NPAD; k <<= 1)
        for (int j = k >> 1; j > 0; j >>= 1)
            bitonic_step_k<<<NPAD / B, B, 0, stream>>>(keys, j, k);
    long othreads = (long)KPOOL * dims[10];
    gather_out_k<<<(int)((othreads + B - 1) / B), B, 0, stream>>>(
        keys, hfin, score, (float*)d_out, KPOOL, dims[10]);
}
